// PointNet2SSGCls_76312978915612
// MI455X (gfx1250) — matrix operand order repaired and checked by reading, not compiled, not run on hardware
//
#include <hip/hip_runtime.h>

// ---------------------------------------------------------------------------
// PointNet++ SSG classification forward for MI455X (gfx1250, wave32, WMMA).
// MLP stages use v_wmma_f32_16x16x32_f16 with b128 fragment loads:
//   - weights pre-swizzled to fragment-major layout (one v16h load per lane)
//   - activations in LDS, A fragments loaded as 2x ds_load_b128 per lane
// ---------------------------------------------------------------------------

typedef __attribute__((ext_vector_type(16))) _Float16 v16h;
typedef __attribute__((ext_vector_type(8)))  _Float16 h8;
typedef __attribute__((ext_vector_type(8)))  float    v8f;

#define BATCH 32
#define NPTS  8192

// ======================= small utility kernels =============================

__global__ void transpose_points_kernel(const float* __restrict__ pts,
                                        float* __restrict__ xyz, int total, int Nn) {
    int i = blockIdx.x * blockDim.x + threadIdx.x;
    if (i >= total) return;
    int b = i / (Nn * 3);
    int rem = i % (Nn * 3);
    int n = rem / 3, c = rem % 3;
    xyz[i] = pts[(size_t)b * 3 * Nn + (size_t)c * Nn + n];
}

// w: [Cout][Cin] f32 -> wF fragment-major f16:
//   element i = ((kt*ntiles + nt)*32 + lane)*16 + h
//   holds w[col=nt*16+(lane&15)][k=kt*32+(lane>>4)*16+h], zero-padded k>=Cin.
// B layout used here: lane -> col=lane&15, K = 16*hi + h (contiguous, same order as A loads).
__global__ void convert_wF_kernel(const float* __restrict__ w, _Float16* __restrict__ wF,
                                  int Cout, int Cin, int Kpad) {
    int i = blockIdx.x * blockDim.x + threadIdx.x;
    int total = Kpad * Cout;
    if (i >= total) return;
    int h = i & 15;
    int lane = (i >> 4) & 31;
    int frag = i >> 9;  // kt*ntiles + nt
    int ntiles = Cout / 16;
    int kt = frag / ntiles;
    int nt = frag % ntiles;
    int k = kt * 32 + (lane >> 4) * 16 + h;
    int col = nt * 16 + (lane & 15);
    float v = (k < Cin) ? w[(size_t)col * Cin + k] : 0.0f;
    wF[i] = (_Float16)v;
}

__global__ void gather_centroids_kernel(const float* __restrict__ xyz,
                                        const int* __restrict__ cidx,
                                        float* __restrict__ out, int total, int Nn, int Mm) {
    int i = blockIdx.x * blockDim.x + threadIdx.x;
    if (i >= total) return;
    int b = i / (Mm * 3);
    int rem = i % (Mm * 3);
    int m = rem / 3, c = rem % 3;
    int pi = cidx[b * Mm + m];
    out[i] = xyz[((size_t)b * Nn + pi) * 3 + c];
}

// ======================= farthest point sampling ===========================
template <int NPT, int NT>
__global__ void fps_kernel(const float* __restrict__ xyz, int n, int m,
                           int* __restrict__ cidx) {
    const int b = blockIdx.x;
    const float* p = xyz + (size_t)b * n * 3;
    const int t = threadIdx.x;

    float px[NPT], py[NPT], pz[NPT], dd[NPT];
#pragma unroll
    for (int j = 0; j < NPT; ++j) {
        int i = t * NPT + j;
        px[j] = p[i * 3 + 0];
        py[j] = p[i * 3 + 1];
        pz[j] = p[i * 3 + 2];
        dd[j] = 1e10f;
    }

    __shared__ float sv[NT];
    __shared__ int   si[NT];

    if (t == 0) cidx[(size_t)b * m] = 0;
    float lx = p[0], ly = p[1], lz = p[2];

    for (int it = 1; it < m; ++it) {
        float bv = -1.0f;
        int   bi = 0;
#pragma unroll
        for (int j = 0; j < NPT; ++j) {
            float dx = px[j] - lx, dy = py[j] - ly, dz = pz[j] - lz;
            float d = dx * dx + dy * dy + dz * dz;
            dd[j] = fminf(dd[j], d);
            if (dd[j] > bv) { bv = dd[j]; bi = t * NPT + j; }
        }
        sv[t] = bv;
        si[t] = bi;
        __syncthreads();
        for (int s = NT / 2; s > 0; s >>= 1) {
            if (t < s) {
                float ov = sv[t + s];
                int   oi = si[t + s];
                if (ov > sv[t] || (ov == sv[t] && oi < si[t])) { sv[t] = ov; si[t] = oi; }
            }
            __syncthreads();
        }
        int nxt = si[0];
        __syncthreads();
        if (t == 0) cidx[(size_t)b * m + it] = nxt;
        lx = p[nxt * 3 + 0];
        ly = p[nxt * 3 + 1];
        lz = p[nxt * 3 + 2];
    }
}

// ============================ ball query ===================================
__global__ void ball_query_kernel(const float* __restrict__ xyz,
                                  const float* __restrict__ new_xyz,
                                  int* __restrict__ idx,
                                  int Bn, int Nn, int Mn, int Kn, float r2) {
    int wavesPerBlock = blockDim.x / 32;
    int w = blockIdx.x * wavesPerBlock + (threadIdx.x / 32);
    int lane = threadIdx.x & 31;
    if (w >= Bn * Mn) return;
    int b = w / Mn, m = w % Mn;

    float cx = new_xyz[(size_t)(b * Mn + m) * 3 + 0];
    float cy = new_xyz[(size_t)(b * Mn + m) * 3 + 1];
    float cz = new_xyz[(size_t)(b * Mn + m) * 3 + 2];
    const float* p = xyz + (size_t)b * Nn * 3;
    int* out = idx + (size_t)w * Kn;

    int cnt = 0, first = 0;
    for (int base = 0; base < Nn && cnt < Kn; base += 32) {
        int i = base + lane;
        float dx = p[i * 3 + 0] - cx;
        float dy = p[i * 3 + 1] - cy;
        float dz = p[i * 3 + 2] - cz;
        bool valid = (dx * dx + dy * dy + dz * dz) < r2;
#if __has_builtin(__builtin_amdgcn_ballot_w32)
        unsigned mask = __builtin_amdgcn_ballot_w32(valid);
#else
        unsigned mask = (unsigned)__ballot(valid);
#endif
        if (mask) {
            if (cnt == 0) first = base + __ffs(mask) - 1;
            if (valid) {
                int pos = __popc(mask & ((1u << lane) - 1u));
                int slot = cnt + pos;
                if (slot < Kn) out[slot] = i;
            }
            cnt += __popc(mask);
        }
    }
    if (cnt > Kn) cnt = Kn;
    for (int s = cnt + lane; s < Kn; s += 32) out[s] = first;
}

// ========================= WMMA MLP building blocks ========================
// Fragment layouts used here (wave32); A and B share a contiguous k order.
// A (16x32 f16): lane<16 -> row=lane,    K = {0..15}
//                lane>=16 -> row=lane-16, K = {16..31}
//   -> one contiguous 16-half run [16*hi, 16*hi+15], read as two h8 loads
// B (32x16 f16): lane -> col=lane&15; K = hi*16 + h (pre-swizzled in wF)
// D (16x16 f32): vgpr r, lane -> m = 8*hi + r, n = lane&15

__device__ __forceinline__ v8f wmma_tile(const _Float16* __restrict__ in, int Cin,
                                         const _Float16* __restrict__ wF, int Cout,
                                         int mt, int nt, int lane) {
    const int n  = lane & 15;
    const int hi = lane >> 4;
    const int row = mt * 16 + n;
    const int ntiles = Cout / 16;
    const v16h* __restrict__ wFv = (const v16h*)wF;
    v8f acc = {};
    for (int kt = 0; kt < Cin / 32; ++kt) {
        const _Float16* ap = in + (size_t)row * Cin + kt * 32 + 16 * hi;
        h8 alo = *(const h8*)(ap);        // k = 16*hi .. 16*hi+7
        h8 ahi = *(const h8*)(ap + 8);    // k = 16*hi+8 .. 16*hi+15
        v16h a = __builtin_shufflevector(alo, ahi, 0, 1, 2, 3, 4, 5, 6, 7,
                                         8, 9, 10, 11, 12, 13, 14, 15);
        v16h bm = wFv[((size_t)kt * ntiles + nt) * 32 + lane];
        acc = __builtin_amdgcn_wmma_f32_16x16x32_f16(false, a, false, bm,
                                                     (short)0, acc, false, false);
    }
    return acc;
}

// One MLP layer (bias+ReLU), f16 output to LDS. Single wave.
template <int KPTS>
__device__ void mlp_layer(const _Float16* __restrict__ in, int Cin,
                          const _Float16* __restrict__ wF, const float* __restrict__ bias,
                          int Cout, _Float16* __restrict__ out) {
    const int lane = threadIdx.x & 31;
    const int n = lane & 15, hi = lane >> 4;
    for (int mt = 0; mt < KPTS / 16; ++mt) {
        for (int nt = 0; nt < Cout / 16; ++nt) {
            v8f acc = wmma_tile(in, Cin, wF, Cout, mt, nt, lane);
            float bv = bias[nt * 16 + n];
#pragma unroll
            for (int r = 0; r < 8; ++r) {
                int m = mt * 16 + hi * 8 + r;
                float v = acc[r] + bv;
                v = v > 0.0f ? v : 0.0f;
                out[(size_t)m * Cout + nt * 16 + n] = (_Float16)v;
            }
        }
    }
    __syncthreads();
}

// Final layer with fused max-over-rows pooling; f32 row of Cout to global.
template <int KPTS>
__device__ void mlp_layer_maxout(const _Float16* __restrict__ in, int Cin,
                                 const _Float16* __restrict__ wF, const float* __restrict__ bias,
                                 int Cout, float* __restrict__ outRow) {
    const int lane = threadIdx.x & 31;
    const int n = lane & 15, hi = lane >> 4;
    for (int nt = 0; nt < Cout / 16; ++nt) {
        float bv = bias[nt * 16 + n];
        float colmax = 0.0f;  // post-ReLU values are >= 0
        for (int mt = 0; mt < KPTS / 16; ++mt) {
            v8f acc = wmma_tile(in, Cin, wF, Cout, mt, nt, lane);
#pragma unroll
            for (int r = 0; r < 8; ++r) {
                float v = acc[r] + bv;
                v = v > 0.0f ? v : 0.0f;
                colmax = fmaxf(colmax, v);
            }
        }
        colmax = fmaxf(colmax, __shfl_xor(colmax, 16, 32));  // combine lane halves
        if (hi == 0) outRow[nt * 16 + n] = colmax;
    }
}

// Final layer, bias+ReLU, f32 output to global (no pooling).
template <int KPTS>
__device__ void mlp_layer_store(const _Float16* __restrict__ in, int Cin,
                                const _Float16* __restrict__ wF, const float* __restrict__ bias,
                                int Cout, float* __restrict__ outG) {
    const int lane = threadIdx.x & 31;
    const int n = lane & 15, hi = lane >> 4;
    for (int mt = 0; mt < KPTS / 16; ++mt) {
        for (int nt = 0; nt < Cout / 16; ++nt) {
            v8f acc = wmma_tile(in, Cin, wF, Cout, mt, nt, lane);
            float bv = bias[nt * 16 + n];
#pragma unroll
            for (int r = 0; r < 8; ++r) {
                int m = mt * 16 + hi * 8 + r;
                float v = acc[r] + bv;
                outG[(size_t)m * Cout + nt * 16 + n] = v > 0.0f ? v : 0.0f;
            }
        }
    }
}

// ==================== SA module: group + 3-layer MLP + max =================
// One wave (32 threads) per centroid.
template <int KPTS, int C0PAD, int C1, int C2, int C3, bool HAS_FEATS, int CF>
__global__ void sa_mlp_kernel(const float* __restrict__ xyz,
                              const float* __restrict__ new_xyz,
                              const int* __restrict__ idx,
                              const float* __restrict__ feats,
                              int Npts, int Mc,
                              const _Float16* __restrict__ w0F, const float* __restrict__ b0,
                              const _Float16* __restrict__ w1F, const float* __restrict__ b1,
                              const _Float16* __restrict__ w2F, const float* __restrict__ b2,
                              float* __restrict__ featsOut) {
    __shared__ __align__(16) _Float16 buf0[KPTS * C0PAD];
    __shared__ __align__(16) _Float16 bufX1[KPTS * C1];
    __shared__ __align__(16) _Float16 bufX2[KPTS * C2];

    const int c = blockIdx.x;
    const int b = c / Mc, m = c % Mc;
    const int lane = threadIdx.x;

    const float cx = new_xyz[(size_t)c * 3 + 0];
    const float cy = new_xyz[(size_t)c * 3 + 1];
    const float cz = new_xyz[(size_t)c * 3 + 2];
    const int* myidx = idx + (size_t)c * KPTS;
    const float* pb = xyz + (size_t)b * Npts * 3;

    // Gather: relative xyz [+ features], pad to C0PAD with zeros, cvt f16.
    for (int e = lane; e < KPTS * C0PAD; e += 32) {
        int kk = e / C0PAD;
        int ch = e % C0PAD;
        int pi = myidx[kk];
        float v = 0.0f;
        if (ch == 0)      v = pb[pi * 3 + 0] - cx;
        else if (ch == 1) v = pb[pi * 3 + 1] - cy;
        else if (ch == 2) v = pb[pi * 3 + 2] - cz;
        else if (HAS_FEATS && (ch - 3) < CF)
            v = feats[((size_t)b * Npts + pi) * CF + (ch - 3)];
        buf0[e] = (_Float16)v;
    }
    __syncthreads();

    mlp_layer<KPTS>(buf0, C0PAD, w0F, b0, C1, bufX1);
    mlp_layer<KPTS>(bufX1, C1, w1F, b1, C2, bufX2);
    mlp_layer_maxout<KPTS>(bufX2, C2, w2F, b2, C3, featsOut + (size_t)c * C3);
}

// =================== loc MLP: rows of 16 points per wave ===================
__global__ void loc_mlp_kernel(const float* __restrict__ feats2,  // [4096][256]
                               const _Float16* __restrict__ w0F, const float* __restrict__ b0,
                               const _Float16* __restrict__ w1F, const float* __restrict__ b1,
                               const _Float16* __restrict__ w2F, const float* __restrict__ b2,
                               float* __restrict__ locout) {      // [4096][1024]
    __shared__ __align__(16) _Float16 buf0[16 * 256];
    __shared__ __align__(16) _Float16 bufX1[16 * 256];
    __shared__ __align__(16) _Float16 bufX2[16 * 512];
    const int r0 = blockIdx.x * 16;
    const int lane = threadIdx.x;

    for (int e = lane; e < 16 * 256; e += 32)
        buf0[e] = (_Float16)feats2[(size_t)r0 * 256 + e];
    __syncthreads();

    mlp_layer<16>(buf0, 256, w0F, b0, 256, bufX1);
    mlp_layer<16>(bufX1, 256, w1F, b1, 512, bufX2);
    mlp_layer_store<16>(bufX2, 512, w2F, b2, 1024, locout + (size_t)r0 * 1024);
}

// global max-pool over `rows` rows per batch
__global__ void maxpool_rows_kernel(const float* __restrict__ in, float* __restrict__ out,
                                    int rows, int cols) {
    int b = blockIdx.x, t = threadIdx.x;
    for (int c = t; c < cols; c += blockDim.x) {
        float mv = in[(size_t)b * rows * cols + c];
        for (int r = 1; r < rows; ++r)
            mv = fmaxf(mv, in[((size_t)b * rows + r) * cols + c]);
        out[(size_t)b * cols + c] = mv;
    }
}

// =================== glob MLP (1024->512->256) + classifier ================
__global__ void glob_cls_kernel(const float* __restrict__ pooled,
                                const float* __restrict__ w0, const float* __restrict__ b0,
                                const float* __restrict__ w1, const float* __restrict__ b1,
                                const float* __restrict__ wc, const float* __restrict__ bc,
                                float* __restrict__ out) {
    __shared__ float x0[1024];
    __shared__ float x1[512];
    __shared__ float x2[256];
    const int b = blockIdx.x, t = threadIdx.x;

    for (int i = t; i < 1024; i += 256) x0[i] = pooled[(size_t)b * 1024 + i];
    __syncthreads();
    for (int o = t; o < 512; o += 256) {
        const float* wr = w0 + (size_t)o * 1024;
        float s = b0[o];
        for (int k = 0; k < 1024; ++k) s += wr[k] * x0[k];
        x1[o] = s > 0.0f ? s : 0.0f;
    }
    __syncthreads();
    for (int o = t; o < 256; o += 256) {
        const float* wr = w1 + (size_t)o * 512;
        float s = b1[o];
        for (int k = 0; k < 512; ++k) s += wr[k] * x1[k];
        x2[o] = s > 0.0f ? s : 0.0f;
    }
    __syncthreads();
    for (int o = t; o < 40; o += 256) {
        const float* wr = wc + (size_t)o * 256;
        float s = bc[o];
        for (int k = 0; k < 256; ++k) s += wr[k] * x2[k];
        out[(size_t)b * 40 + o] = s;
    }
}

// ============================ host launcher ================================

extern "C" void kernel_launch(void* const* d_in, const int* in_sizes, int n_in,
                              void* d_out, int out_size, void* d_ws, size_t ws_size,
                              hipStream_t stream) {
    (void)in_sizes; (void)n_in; (void)out_size; (void)ws_size;

    const float* points = (const float*)d_in[0];
    const float* sa1_w[3] = {(const float*)d_in[1], (const float*)d_in[3], (const float*)d_in[5]};
    const float* sa1_b[3] = {(const float*)d_in[2], (const float*)d_in[4], (const float*)d_in[6]};
    const float* sa2_w[3] = {(const float*)d_in[7], (const float*)d_in[9], (const float*)d_in[11]};
    const float* sa2_b[3] = {(const float*)d_in[8], (const float*)d_in[10], (const float*)d_in[12]};
    const float* loc_w[3] = {(const float*)d_in[13], (const float*)d_in[15], (const float*)d_in[17]};
    const float* loc_b[3] = {(const float*)d_in[14], (const float*)d_in[16], (const float*)d_in[18]};
    const float* glob_w0 = (const float*)d_in[19];
    const float* glob_b0 = (const float*)d_in[20];
    const float* glob_w1 = (const float*)d_in[21];
    const float* glob_b1 = (const float*)d_in[22];
    const float* cls_w = (const float*)d_in[23];
    const float* cls_b = (const float*)d_in[24];
    float* out = (float*)d_out;

    // ---- workspace bump allocator (256B aligned) ----
    char* ws = (char*)d_ws;
    size_t off = 0;
    auto alloc = [&](size_t bytes) -> void* {
        void* p = ws + off;
        off += (bytes + 255) & ~(size_t)255;
        return p;
    };

    float* xyz    = (float*)alloc((size_t)BATCH * NPTS * 3 * 4);
    int*   cidx1  = (int*)  alloc((size_t)BATCH * 512 * 4);
    float* nx1    = (float*)alloc((size_t)BATCH * 512 * 3 * 4);
    int*   idx1   = (int*)  alloc((size_t)BATCH * 512 * 32 * 4);
    float* feats1 = (float*)alloc((size_t)BATCH * 512 * 128 * 4);
    int*   cidx2  = (int*)  alloc((size_t)BATCH * 128 * 4);
    float* nx2    = (float*)alloc((size_t)BATCH * 128 * 3 * 4);
    int*   idx2   = (int*)  alloc((size_t)BATCH * 128 * 64 * 4);
    float* feats2 = (float*)alloc((size_t)BATCH * 128 * 256 * 4);
    float* locout = (float*)alloc((size_t)BATCH * 128 * 1024 * 4);
    float* pooled = (float*)alloc((size_t)BATCH * 1024 * 4);

    _Float16* sa1F[3] = {(_Float16*)alloc(32 * 64 * 2),  (_Float16*)alloc(64 * 64 * 2),
                         (_Float16*)alloc(64 * 128 * 2)};
    _Float16* sa2F[3] = {(_Float16*)alloc(160 * 128 * 2), (_Float16*)alloc(128 * 128 * 2),
                         (_Float16*)alloc(128 * 256 * 2)};
    _Float16* locF[3] = {(_Float16*)alloc(256 * 256 * 2), (_Float16*)alloc(256 * 512 * 2),
                         (_Float16*)alloc(512 * 1024 * 2)};

    // ---- weight conversion: [Cout][Cin] f32 -> fragment-major f16 ----
    struct WCvt { const float* w; _Float16* wF; int Cout, Cin, Kpad; };
    WCvt cv[9] = {
        {sa1_w[0], sa1F[0], 64, 3, 32},    {sa1_w[1], sa1F[1], 64, 64, 64},
        {sa1_w[2], sa1F[2], 128, 64, 64},  {sa2_w[0], sa2F[0], 128, 131, 160},
        {sa2_w[1], sa2F[1], 128, 128, 128},{sa2_w[2], sa2F[2], 256, 128, 128},
        {loc_w[0], locF[0], 256, 256, 256},{loc_w[1], locF[1], 512, 256, 256},
        {loc_w[2], locF[2], 1024, 512, 512}};
    for (int i = 0; i < 9; ++i) {
        int total = cv[i].Kpad * cv[i].Cout;
        convert_wF_kernel<<<(total + 255) / 256, 256, 0, stream>>>(
            cv[i].w, cv[i].wF, cv[i].Cout, cv[i].Cin, cv[i].Kpad);
    }

    // ---- [B,3,N] -> [B,N,3] ----
    {
        int total = BATCH * NPTS * 3;
        transpose_points_kernel<<<(total + 255) / 256, 256, 0, stream>>>(points, xyz, total, NPTS);
    }

    // ---- SA1: FPS 8192->512, ball query r=0.2 k=32, MLP 3->64->64->128 ----
    fps_kernel<8, 1024><<<BATCH, 1024, 0, stream>>>(xyz, NPTS, 512, cidx1);
    {
        int total = BATCH * 512 * 3;
        gather_centroids_kernel<<<(total + 255) / 256, 256, 0, stream>>>(xyz, cidx1, nx1, total, NPTS, 512);
    }
    ball_query_kernel<<<(BATCH * 512) / 8, 256, 0, stream>>>(xyz, nx1, idx1, BATCH, NPTS, 512, 32, 0.04f);
    sa_mlp_kernel<32, 32, 64, 64, 128, false, 0><<<BATCH * 512, 32, 0, stream>>>(
        xyz, nx1, idx1, nullptr, NPTS, 512,
        sa1F[0], sa1_b[0], sa1F[1], sa1_b[1], sa1F[2], sa1_b[2], feats1);

    // ---- SA2: FPS 512->128, ball query r=0.4 k=64, MLP 131->128->128->256 ----
    fps_kernel<1, 512><<<BATCH, 512, 0, stream>>>(nx1, 512, 128, cidx2);
    {
        int total = BATCH * 128 * 3;
        gather_centroids_kernel<<<(total + 255) / 256, 256, 0, stream>>>(nx1, cidx2, nx2, total, 512, 128);
    }
    ball_query_kernel<<<(BATCH * 128) / 8, 256, 0, stream>>>(nx1, nx2, idx2, BATCH, 512, 128, 64, 0.16f);
    sa_mlp_kernel<64, 160, 128, 128, 256, true, 128><<<BATCH * 128, 32, 0, stream>>>(
        nx1, nx2, idx2, feats1, 512, 128,
        sa2F[0], sa2_b[0], sa2F[1], sa2_b[1], sa2F[2], sa2_b[2], feats2);

    // ---- loc MLP 256->256->512->1024 on 4096 rows, then max over 128 pts ----
    loc_mlp_kernel<<<(BATCH * 128) / 16, 32, 0, stream>>>(
        feats2, locF[0], loc_b[0], locF[1], loc_b[1], locF[2], loc_b[2], locout);
    maxpool_rows_kernel<<<BATCH, 256, 0, stream>>>(locout, pooled, 128, 1024);

    // ---- glob MLP + classifier ----
    glob_cls_kernel<<<BATCH, 256, 0, stream>>>(pooled, glob_w0, glob_b0, glob_w1, glob_b1,
                                               cls_w, cls_b, out);
}
